// PointNextPatchTokenizer_50500225466737
// MI455X (gfx1250) — matrix operand order repaired and checked by reading, not compiled, not run on hardware
//
#include <hip/hip_runtime.h>
#include <hip/hip_bf16.h>
#include <math.h>

typedef __attribute__((ext_vector_type(16))) _Float16 v16h;
typedef __attribute__((ext_vector_type(8)))  _Float16 v8h;
typedef __attribute__((ext_vector_type(8)))  float    v8f;

#define B_    16
#define N_    16384
#define INC   6
#define STEMC 64
#define TOKC  128
#define NP    64

// ---------- wave-level helpers (ds_swizzle xor-lane ops) ------------------

template <int MASK>
__device__ __forceinline__ float xor_val(float v) {
  int x = __builtin_amdgcn_ds_swizzle(__float_as_int(v), 0x1f | (MASK << 10));
  return __int_as_float(x);
}
template <int MASK>
__device__ __forceinline__ int xor_vali(int v) {
  return __builtin_amdgcn_ds_swizzle(v, 0x1f | (MASK << 10));
}

__device__ __forceinline__ float half16_sum(float v) {
  v += xor_val<1>(v);
  v += xor_val<2>(v);
  v += xor_val<4>(v);
  v += xor_val<8>(v);
  return v;
}

template <int MASK>
__device__ __forceinline__ void argmax_step(float& bv, int& bi) {
  float ov = xor_val<MASK>(bv);
  int   oi = xor_vali<MASK>(bi);
  if (ov > bv || (ov == bv && oi < bi)) { bv = ov; bi = oi; }
}

__device__ __forceinline__ v8f zero8() {
  v8f z;
#pragma unroll
  for (int i = 0; i < 8; ++i) z[i] = 0.0f;
  return z;
}

__device__ __forceinline__ v8f wmma16(v16h a, v16h b, v8f c) {
  return __builtin_amdgcn_wmma_f32_16x16x32_f16(false, a, false, b,
                                                (short)0, c, false, false);
}

// Pre-swizzled B fragment: one unconditional 32B load per lane.
__device__ __forceinline__ v16h load_bfrag(const _Float16* __restrict__ wp,
                                           int nt, int kc, int t, int lane) {
  return *(const v16h*)(wp + (((size_t)(kc * nt + t) * 32 + lane) << 4));
}

// A fragment from LDS staging: two 16B ds loads (always 16B-aligned here).
__device__ __forceinline__ v16h load_afrag(const _Float16* s, int ldk,
                                           int kbase, int lane) {
  const _Float16* row = s + (size_t)(lane & 15) * ldk;
  int klo = kbase + ((lane & 16) ? 8 : 0);
  v8h lo = *(const v8h*)(row + klo);
  v8h hi = *(const v8h*)(row + klo + 16);
  v16h a;
#pragma unroll
  for (int i = 0; i < 8; ++i) { a[i] = lo[i]; a[8 + i] = hi[i]; }
  return a;
}

// bias + LayerNorm (+optional exact gelu) over T accumulator tiles (C = 16*T).
template <int T, bool ACT>
__device__ __forceinline__ void bias_ln_act(v8f* acc, const float* __restrict__ bias,
                                            const float* __restrict__ g,
                                            const float* __restrict__ be, int lane) {
  const float C = 16.0f * (float)T;
  float bv[T], gv[T], bev[T];
#pragma unroll
  for (int t = 0; t < T; ++t) {
    int ch = 16 * t + (lane & 15);
    bv[t] = bias[ch]; gv[t] = g[ch]; bev[t] = be[ch];
  }
#pragma unroll
  for (int r = 0; r < 8; ++r) {
    float s = 0.0f, ss = 0.0f;
#pragma unroll
    for (int t = 0; t < T; ++t) {
      float x = acc[t][r] + bv[t];
      acc[t][r] = x; s += x; ss += x * x;
    }
    s  = half16_sum(s);
    ss = half16_sum(ss);
    float mean = s / C;
    float var  = ss / C - mean * mean;
    float rstd = rsqrtf(var + 1e-5f);
#pragma unroll
    for (int t = 0; t < T; ++t) {
      float x = (acc[t][r] - mean) * rstd * gv[t] + bev[t];
      if (ACT) x = x * 0.5f * (1.0f + erff(x * 0.70710678118654752f));
      acc[t][r] = x;
    }
  }
}

// ---------- kernel 0: swizzle f32 weights into B-fragment f16 layout ------

__global__ __launch_bounds__(256) void k_wprep(const float* __restrict__ w,
                                               _Float16* __restrict__ dst,
                                               int cin, int cout, int nkc) {
  int nt = cout >> 4;
  int total = nkc * nt * 512;
  int idx = blockIdx.x * 256 + threadIdx.x;
  if (idx >= total) return;
  int i    = idx & 15;
  int lane = (idx >> 4) & 31;
  int ft   = idx >> 9;               // kc*nt + t
  int t    = ft % nt;
  int n    = t * 16 + (lane & 15);
  int kc   = ft / nt;
  int k    = kc * 32 + ((lane & 16) ? 8 : 0) + ((i < 8) ? i : (8 + i));
  dst[idx] = (k < cin) ? (_Float16)w[(size_t)k * cout + n] : (_Float16)0.0f;
}

// ---------- kernel 1: stem MLP 6 -> 64 -> 64 over all points --------------

__global__ __launch_bounds__(256) void k_stem(
    const float* __restrict__ pc,
    const _Float16* __restrict__ wp0, const float* __restrict__ b0,
    const float* __restrict__ g0, const float* __restrict__ be0,
    const _Float16* __restrict__ wp1, const float* __restrict__ b1,
    const float* __restrict__ g1, const float* __restrict__ be1,
    _Float16* __restrict__ stemf) {
  __shared__ _Float16 stg[8][16 * STEMC];
  int wid  = threadIdx.x >> 5;
  int lane = threadIdx.x & 31;
  int p0   = blockIdx.x * 128 + wid * 16;   // 16 points per wave

  // A fragment for layer0: K = 6 real inputs, rest zero (only lanes<16 carry data)
  v16h a0;
#pragma unroll
  for (int i = 0; i < 16; ++i) a0[i] = (_Float16)0.0f;
  if (lane < 16) {
    const float* p = pc + (size_t)(p0 + lane) * INC;
#pragma unroll
    for (int i = 0; i < INC; ++i) a0[i] = (_Float16)p[i];
  }

  v8f acc[4];
#pragma unroll
  for (int t = 0; t < 4; ++t) acc[t] = zero8();
#pragma unroll
  for (int t = 0; t < 4; ++t)
    acc[t] = wmma16(a0, load_bfrag(wp0, 4, 0, t, lane), acc[t]);
  bias_ln_act<4, true>(acc, b0, g0, be0, lane);

  int rowoff = (lane & 16) ? 8 : 0;
#pragma unroll
  for (int t = 0; t < 4; ++t)
#pragma unroll
    for (int r = 0; r < 8; ++r)
      stg[wid][(r + rowoff) * STEMC + 16 * t + (lane & 15)] = (_Float16)acc[t][r];
  __syncthreads();

  v16h a1_0 = load_afrag(stg[wid], STEMC, 0,  lane);
  v16h a1_1 = load_afrag(stg[wid], STEMC, 32, lane);
  v8f acc2[4];
#pragma unroll
  for (int t = 0; t < 4; ++t) acc2[t] = zero8();
#pragma unroll
  for (int t = 0; t < 4; ++t) {
    acc2[t] = wmma16(a1_0, load_bfrag(wp1, 4, 0, t, lane), acc2[t]);
    acc2[t] = wmma16(a1_1, load_bfrag(wp1, 4, 1, t, lane), acc2[t]);
  }
  bias_ln_act<4, true>(acc2, b1, g1, be1, lane);
  __syncthreads();                            // stg reads done, safe to rewrite

#pragma unroll
  for (int t = 0; t < 4; ++t)
#pragma unroll
    for (int r = 0; r < 8; ++r)
      stg[wid][(r + rowoff) * STEMC + 16 * t + (lane & 15)] = (_Float16)acc2[t][r];
  __syncthreads();

  // coalesced 16B-chunk store of the 16x64 f16 tile (contiguous in stemf)
  _Float16* dstt = stemf + (size_t)p0 * STEMC;
#pragma unroll
  for (int c = 0; c < 4; ++c) {
    int flat = lane * 4 + c;                  // 128 chunks of 8 halfs
    *(v8h*)(dstt + flat * 8) = *(const v8h*)(stg[wid] + flat * 8);
  }
}

// ---------- kernel 2: farthest point sampling (one block per batch) -------

__global__ __launch_bounds__(1024) void k_fps(const float* __restrict__ pc,
                                              int* __restrict__ cidx,
                                              float* __restrict__ centers) {
  extern __shared__ float lds[];
  float* sx   = lds;
  float* sy   = sx + N_;
  float* sz   = sy + N_;
  float* dist = sz + N_;
  __shared__ float rbv[32];
  __shared__ int   rbi[32];
  __shared__ int   sfar;

  int b = blockIdx.x, tid = threadIdx.x;
  for (int i = tid; i < N_; i += 1024) {
    const float* p = pc + (size_t)(b * N_ + i) * INC;
    sx[i] = p[0]; sy[i] = p[1]; sz[i] = p[2];
    dist[i] = 1e10f;
  }
  __syncthreads();

  int far = 0;
  for (int it = 0; it < NP; ++it) {
    if (tid == 0) {
      cidx[b * NP + it] = far;
      centers[(size_t)(b * NP + it) * 3 + 0] = sx[far];
      centers[(size_t)(b * NP + it) * 3 + 1] = sy[far];
      centers[(size_t)(b * NP + it) * 3 + 2] = sz[far];
    }
    float cx = sx[far], cy = sy[far], cz = sz[far];
    float bv = -1.0f; int bi = 0;
    for (int i = tid; i < N_; i += 1024) {
      float dx = sx[i] - cx, dy = sy[i] - cy, dz = sz[i] - cz;
      float d = fminf(dist[i], dx * dx + dy * dy + dz * dz);
      dist[i] = d;
      if (d > bv) { bv = d; bi = i; }
    }
    argmax_step<1>(bv, bi);
    argmax_step<2>(bv, bi);
    argmax_step<4>(bv, bi);
    argmax_step<8>(bv, bi);
    argmax_step<16>(bv, bi);
    if ((tid & 31) == 0) { rbv[tid >> 5] = bv; rbi[tid >> 5] = bi; }
    __syncthreads();
    if (tid < 32) {
      bv = rbv[tid]; bi = rbi[tid];
      argmax_step<1>(bv, bi);
      argmax_step<2>(bv, bi);
      argmax_step<4>(bv, bi);
      argmax_step<8>(bv, bi);
      argmax_step<16>(bv, bi);
      if (tid == 0) sfar = bi;
    }
    __syncthreads();
    far = sfar;
  }
}

// ---------- kernel 3: ball query (one wave per center, ballot compaction) -

__global__ __launch_bounds__(256) void k_ballq(const float* __restrict__ pc,
                                               const float* __restrict__ centers,
                                               int* __restrict__ nidx0,
                                               int* __restrict__ nidx1) {
  __shared__ int lbig[8][32];
  __shared__ int lsml[8][16];
  int wid  = threadIdx.x >> 5;
  int lane = threadIdx.x & 31;
  int wc   = blockIdx.x * 8 + wid;
  int b    = wc >> 6;
  float cx = centers[(size_t)wc * 3 + 0];
  float cy = centers[(size_t)wc * 3 + 1];
  float cz = centers[(size_t)wc * 3 + 2];
  const float r0sq = 0.04f * 0.04f, r1sq = 0.08f * 0.08f;

  int cntB = 0, cntS = 0;
  for (int base = 0; base < N_ && (cntB < 32 || cntS < 16); base += 32) {
    int i = base + lane;
    const float* p = pc + (size_t)(b * N_ + i) * INC;
    float dx = p[0] - cx, dy = p[1] - cy, dz = p[2] - cz;
    float d = dx * dx + dy * dy + dz * dz;
    bool inB = d <= r1sq, inS = d <= r0sq;
    unsigned long long mB = __ballot(inB);
    unsigned long long mS = __ballot(inS);
    unsigned long long lm = (lane == 0) ? 0ull : ((~0ull) >> (64 - lane));
    if (inB) {
      int slot = cntB + __popcll(mB & lm);
      if (slot < 32) lbig[wid][slot] = i;
    }
    if (inS) {
      int slot = cntS + __popcll(mS & lm);
      if (slot < 16) lsml[wid][slot] = i;
    }
    cntB += __popcll(mB);
    cntS += __popcll(mS);
  }
  cntB = min(cntB, 32);
  cntS = min(cntS, 16);
  int firstB = lbig[wid][0];
  int firstS = (cntS > 0) ? lsml[wid][0] : firstB;
  nidx1[(size_t)wc * 32 + lane] = (lane < cntB) ? lbig[wid][lane] : firstB;
  if (lane < 16)
    nidx0[(size_t)wc * 16 + lane] = (lane < cntS) ? lsml[wid][lane] : firstS;
}

// ---------- kernel 4: grouped encoder MLP 91->128->128 + max pool ---------

__global__ __launch_bounds__(128) void k_enc(
    const _Float16* __restrict__ stemf, const float* __restrict__ pc,
    const float* __restrict__ centers, const int* __restrict__ nidx,
    const _Float16* __restrict__ wp0, const float* __restrict__ b0,
    const float* __restrict__ g0, const float* __restrict__ be0,
    const _Float16* __restrict__ wp1, const float* __restrict__ b1,
    const float* __restrict__ g1, const float* __restrict__ be1,
    _Float16* __restrict__ sfeat, int k, float invr) {
  __shared__ _Float16 gstg[4][16 * 96];
  __shared__ _Float16 hstg[4][16 * TOKC];
  const float freq4[4] = {1.0f, 0.1f, 0.01f, 0.001f};

  int wid  = threadIdx.x >> 5;
  int lane = threadIdx.x & 31;
  int wc   = blockIdx.x * 4 + wid;
  int b    = wc >> 6;
  float ctr[3];
#pragma unroll
  for (int d = 0; d < 3; ++d) ctr[d] = centers[(size_t)wc * 3 + d];

  float maxv[8];
#pragma unroll
  for (int t = 0; t < 8; ++t) maxv[t] = -3.0e38f;

  int nb = k >> 4;
  for (int jb = 0; jb < nb; ++jb) {
    int row = lane & 15;
    int pi  = nidx[(size_t)wc * k + jb * 16 + row];
    _Float16* grow = &gstg[wid][row * 96];
    const v8h* sfv = (const v8h*)(stemf + (size_t)(b * N_ + pi) * STEMC);
    if (lane < 16) {                              // nfeat halves 0..47 (6 x 16B)
#pragma unroll
      for (int c = 0; c < 6; ++c) *(v8h*)(grow + c * 8) = sfv[c];
    } else {                                      // nfeat 48..63 + rel + pe + pad
      *(v8h*)(grow + 48) = sfv[6];
      *(v8h*)(grow + 56) = sfv[7];
      const float* p = pc + (size_t)(b * N_ + pi) * INC;
      float rel[3];
#pragma unroll
      for (int d = 0; d < 3; ++d) rel[d] = (p[d] - ctr[d]) * invr;
#pragma unroll
      for (int d = 0; d < 3; ++d) grow[64 + d] = (_Float16)rel[d];
#pragma unroll
      for (int d = 0; d < 3; ++d)
#pragma unroll
        for (int j = 0; j < 4; ++j) {
          float ang = rel[d] * freq4[j];
          grow[67 + d * 8 + j]     = (_Float16)__sinf(ang);
          grow[67 + d * 8 + 4 + j] = (_Float16)__cosf(ang);
        }
#pragma unroll
      for (int kk = 91; kk < 96; ++kk) grow[kk] = (_Float16)0.0f;
    }
    __syncthreads();

    v16h af[3];
#pragma unroll
    for (int kc = 0; kc < 3; ++kc) af[kc] = load_afrag(gstg[wid], 96, kc * 32, lane);

    v8f acc[8];
#pragma unroll
    for (int t = 0; t < 8; ++t) acc[t] = zero8();
#pragma unroll
    for (int kc = 0; kc < 3; ++kc)
#pragma unroll
      for (int t = 0; t < 8; ++t)
        acc[t] = wmma16(af[kc], load_bfrag(wp0, 8, kc, t, lane), acc[t]);
    bias_ln_act<8, true>(acc, b0, g0, be0, lane);

    int rowoff = (lane & 16) ? 8 : 0;
#pragma unroll
    for (int t = 0; t < 8; ++t)
#pragma unroll
      for (int r = 0; r < 8; ++r)
        hstg[wid][(r + rowoff) * TOKC + 16 * t + (lane & 15)] = (_Float16)acc[t][r];
    __syncthreads();

    v8f acc2[8];
#pragma unroll
    for (int t = 0; t < 8; ++t) acc2[t] = zero8();
#pragma unroll
    for (int kc = 0; kc < 4; ++kc) {
      v16h a = load_afrag(hstg[wid], TOKC, kc * 32, lane);
#pragma unroll
      for (int t = 0; t < 8; ++t)
        acc2[t] = wmma16(a, load_bfrag(wp1, 8, kc, t, lane), acc2[t]);
    }
    bias_ln_act<8, false>(acc2, b1, g1, be1, lane);

#pragma unroll
    for (int t = 0; t < 8; ++t)
#pragma unroll
      for (int r = 0; r < 8; ++r) maxv[t] = fmaxf(maxv[t], acc2[t][r]);
    __syncthreads();
  }

#pragma unroll
  for (int t = 0; t < 8; ++t) {
    float m = fmaxf(maxv[t], xor_val<16>(maxv[t]));
    if (lane < 16) sfeat[(size_t)wc * TOKC + 16 * t + lane] = (_Float16)m;
  }
}

// ---------- kernel 5: token MLP 416 -> 128 -> 128 -------------------------

__global__ __launch_bounds__(128) void k_tp(
    const _Float16* __restrict__ stemf, const int* __restrict__ cidx,
    const float* __restrict__ centers, const _Float16* __restrict__ s0,
    const _Float16* __restrict__ s1,
    const _Float16* __restrict__ wp0, const float* __restrict__ b0,
    const float* __restrict__ g0, const float* __restrict__ be0,
    const _Float16* __restrict__ wp1, const float* __restrict__ b1,
    const float* __restrict__ g1, const float* __restrict__ be1,
    float* __restrict__ out) {
  __shared__ _Float16 tstg[4][16 * 416];
  // 10000^(-j/16), j = 0..15
  const float freq16[16] = {
      1.0f, 0.5623413252f, 0.3162277660f, 0.1778279410f,
      0.1f, 0.0562341325f, 0.0316227766f, 0.0177827941f,
      0.01f, 0.0056234133f, 0.0031622777f, 0.0017782794f,
      0.001f, 0.0005623413f, 0.0003162278f, 0.0001778279f};
  int wid  = threadIdx.x >> 5;
  int lane = threadIdx.x & 31;
  int gw   = blockIdx.x * 4 + wid;
  int tokbase = gw * 16;
  int row  = lane & 15;
  int tok  = tokbase + row;
  int b    = tok >> 6;

  _Float16* trow = &tstg[wid][row * 416];
  if (lane < 16) {                       // K 0..207
    int pi = cidx[tok];
    const v8h* sfv = (const v8h*)(stemf + (size_t)(b * N_ + pi) * STEMC);
#pragma unroll
    for (int c = 0; c < 8; ++c) *(v8h*)(trow + c * 8) = sfv[c];
    const v8h* f0 = (const v8h*)(s0 + (size_t)tok * TOKC);
#pragma unroll
    for (int c = 0; c < 16; ++c) *(v8h*)(trow + 64 + c * 8) = f0[c];
    const v8h* f1 = (const v8h*)(s1 + (size_t)tok * TOKC);
    *(v8h*)(trow + 192) = f1[0];
    *(v8h*)(trow + 200) = f1[1];
  } else {                               // K 208..415
    const v8h* f1 = (const v8h*)(s1 + (size_t)tok * TOKC);
#pragma unroll
    for (int c = 2; c < 16; ++c) *(v8h*)(trow + 192 + c * 8) = f1[c];
#pragma unroll
    for (int d = 0; d < 3; ++d) {
      float c = centers[(size_t)tok * 3 + d];
#pragma unroll
      for (int j = 0; j < 16; ++j) {
        float ang = c * freq16[j];
        trow[320 + d * 32 + j]      = (_Float16)__sinf(ang);
        trow[320 + d * 32 + 16 + j] = (_Float16)__cosf(ang);
      }
    }
  }
  __syncthreads();

  v8f acc[8];
#pragma unroll
  for (int t = 0; t < 8; ++t) acc[t] = zero8();
  for (int kc = 0; kc < 13; ++kc) {
    v16h a = load_afrag(tstg[wid], 416, kc * 32, lane);
#pragma unroll
    for (int t = 0; t < 8; ++t)
      acc[t] = wmma16(a, load_bfrag(wp0, 8, kc, t, lane), acc[t]);
  }
  bias_ln_act<8, true>(acc, b0, g0, be0, lane);
  __syncthreads();

  int rowoff = (lane & 16) ? 8 : 0;
#pragma unroll
  for (int t = 0; t < 8; ++t)
#pragma unroll
    for (int r = 0; r < 8; ++r)
      tstg[wid][(r + rowoff) * 416 + 16 * t + (lane & 15)] = (_Float16)acc[t][r];
  __syncthreads();

  v8f acc2[8];
#pragma unroll
  for (int t = 0; t < 8; ++t) acc2[t] = zero8();
#pragma unroll
  for (int kc = 0; kc < 4; ++kc) {
    v16h a = load_afrag(tstg[wid], 416, kc * 32, lane);
#pragma unroll
    for (int t = 0; t < 8; ++t)
      acc2[t] = wmma16(a, load_bfrag(wp1, 8, kc, t, lane), acc2[t]);
  }
  bias_ln_act<8, false>(acc2, b1, g1, be1, lane);

#pragma unroll
  for (int t = 0; t < 8; ++t)
#pragma unroll
    for (int r = 0; r < 8; ++r)
      out[(size_t)(tokbase + r + rowoff) * TOKC + 16 * t + (lane & 15)] = acc2[t][r];
}

// ---------- host launcher -------------------------------------------------

extern "C" void kernel_launch(void* const* d_in, const int* in_sizes, int n_in,
                              void* d_out, int out_size, void* d_ws, size_t ws_size,
                              hipStream_t stream) {
  (void)in_sizes; (void)n_in; (void)out_size; (void)ws_size;
  const float* pc = (const float*)d_in[0];
  const float* P[33];
  for (int i = 1; i <= 32; ++i) P[i] = (const float*)d_in[i];

  float* out     = (float*)d_out;
  float* centers = out + (size_t)B_ * NP * TOKC;

  char* ws = (char*)d_ws;
  size_t off = 0;
  auto take = [&](size_t bytes) {
    char* p = ws + off;
    off = (off + bytes + 255) & ~(size_t)255;
    return p;
  };
  _Float16* stemf = (_Float16*)take((size_t)B_ * N_ * STEMC * 2);
  int* cidx       = (int*)take((size_t)B_ * NP * 4);
  int* nidx0      = (int*)take((size_t)B_ * NP * 16 * 4);
  int* nidx1      = (int*)take((size_t)B_ * NP * 32 * 4);
  _Float16* s0f   = (_Float16*)take((size_t)B_ * NP * TOKC * 2);
  _Float16* s1f   = (_Float16*)take((size_t)B_ * NP * TOKC * 2);
  // pre-swizzled weights: [nkc][ntile][32 lanes][16 halfs]
  _Float16* wps0  = (_Float16*)take(1  * 4 * 512 * 2);  // stem0  6->64
  _Float16* wps1  = (_Float16*)take(2  * 4 * 512 * 2);  // stem1  64->64
  _Float16* wpe00 = (_Float16*)take(3  * 8 * 512 * 2);  // enc0_0 91->128
  _Float16* wpe01 = (_Float16*)take(4  * 8 * 512 * 2);  // enc0_1 128->128
  _Float16* wpe10 = (_Float16*)take(3  * 8 * 512 * 2);  // enc1_0 91->128
  _Float16* wpe11 = (_Float16*)take(4  * 8 * 512 * 2);  // enc1_1 128->128
  _Float16* wpt0  = (_Float16*)take(13 * 8 * 512 * 2);  // tp0    416->128
  _Float16* wpt1  = (_Float16*)take(4  * 8 * 512 * 2);  // tp1    128->128

  auto prep = [&](const float* w, _Float16* dst, int cin, int cout, int nkc) {
    int total = nkc * (cout >> 4) * 512;
    k_wprep<<<(total + 255) / 256, 256, 0, stream>>>(w, dst, cin, cout, nkc);
  };
  prep(P[1],  wps0,  INC,   STEMC, 1);
  prep(P[5],  wps1,  STEMC, STEMC, 2);
  prep(P[9],  wpe00, 91,    TOKC,  3);
  prep(P[13], wpe01, TOKC,  TOKC,  4);
  prep(P[17], wpe10, 91,    TOKC,  3);
  prep(P[21], wpe11, TOKC,  TOKC,  4);
  prep(P[25], wpt0,  416,   TOKC,  13);
  prep(P[29], wpt1,  TOKC,  TOKC,  4);

  // 1) stem MLP over all points (WMMA)
  k_stem<<<(B_ * N_) / 128, 256, 0, stream>>>(pc,
      wps0, P[2], P[3], P[4], wps1, P[6], P[7], P[8], stemf);

  // 2) FPS: xyz + dist resident in 256KB dynamic LDS (gfx1250 320KB WGP LDS)
  size_t fps_lds = (size_t)N_ * 4 * sizeof(float);
  k_fps<<<B_, 1024, fps_lds, stream>>>(pc, cidx, centers);

  // 3) ball query, both radii in one scan pass per center
  k_ballq<<<(B_ * NP) / 8, 256, 0, stream>>>(pc, centers, nidx0, nidx1);

  // 4) encoder MLPs per scale (WMMA) with neighbor max-pool
  k_enc<<<(B_ * NP) / 4, 128, 0, stream>>>(stemf, pc, centers, nidx0,
      wpe00, P[10], P[11], P[12], wpe01, P[14], P[15], P[16], s0f, 16, 25.0f);
  k_enc<<<(B_ * NP) / 4, 128, 0, stream>>>(stemf, pc, centers, nidx1,
      wpe10, P[18], P[19], P[20], wpe11, P[22], P[23], P[24], s1f, 32, 12.5f);

  // 5) token MLP (WMMA), writes patch_token f32 directly to d_out
  k_tp<<<(B_ * NP) / 64, 128, 0, stream>>>(stemf, cidx, centers, s0f, s1f,
      wpt0, P[26], P[27], P[28], wpt1, P[30], P[31], P[32], out);
}
